// AttentionLayer_1683627180613
// MI455X (gfx1250) — compile-verified, matrix operand order checked by reading, not run on hardware
//
#include <hip/hip_runtime.h>

// Problem constants (match reference)
#define Bsz 8
#define TE  512
#define TD  256
#define Hd  256
#define JT  4      // decoder positions per energy block

typedef __attribute__((ext_vector_type(16))) __bf16 v16bf;
typedef __attribute__((ext_vector_type(8)))  float  v8f;

union Frag { uint4 u[2]; v16bf v; };

// Hardware tanh on the trans pipe (V_TANH_F32 is new on gfx1250).
__device__ __forceinline__ float fast_tanh(float x)
{
#if __has_builtin(__builtin_amdgcn_tanhf)
    return __builtin_amdgcn_tanhf(x);
#else
    // 2 trans ops (v_exp_f32 + v_rcp_f32) — still far cheaper than OCML tanh
    const float e = __expf(2.0f * x);
    return 1.0f - 2.0f / (e + 1.0f);
#endif
}

// ---------------------------------------------------------------------------
// Generic bf16 WMMA GEMM: D[M,N](f32) = A[M,K](bf16, row-major) * B[K,N]
// where B is supplied pre-transposed: Bt[N,K] row-major.
// One wave -> one 16x16 D tile; 4 waves/block cover 64 columns.
// Fragment layout per CDNA5 ISA 7.12.2 (16-bit A, wave32):
//   lanes 0-15 : row M=lane,    elems 0..7 = K0..7,  8..15 = K16..23
//   lanes 16-31: row M=lane-16, elems 0..7 = K8..15, 8..15 = K24..31
// -> two aligned 16B loads per operand per K-step of 32.
// ---------------------------------------------------------------------------
__global__ __launch_bounds__(128)
void gemm_bf16_wmma(const __bf16* __restrict__ A, const __bf16* __restrict__ Bt,
                    float* __restrict__ D, int M, int N, int K,
                    int lda, int ldb, int ldc,
                    long sA, long sB, long sD)
{
    const int z = blockIdx.z;
    A  += (long)z * sA;
    Bt += (long)z * sB;
    D  += (long)z * sD;

    const int wave = threadIdx.x >> 5;
    const int lane = threadIdx.x & 31;
    const int m0 = blockIdx.x * 16;
    const int n0 = blockIdx.y * 64 + wave * 16;
    if (m0 >= M || n0 >= N) return;

    const int r  = lane & 15;            // row (A) / col (B) owned by this lane
    const int kb = (lane >> 4) << 3;     // K sub-offset: 0 for lanes 0-15, 8 for 16-31

    const __bf16* arow = A  + (long)(m0 + r) * lda + kb;
    const __bf16* brow = Bt + (long)(n0 + r) * ldb + kb;

    v8f acc = {};
    for (int k0 = 0; k0 < K; k0 += 32) {
        Frag fa, fb;
        fa.u[0] = *(const uint4*)(arow + k0);        // K = k0+kb .. +7
        fa.u[1] = *(const uint4*)(arow + k0 + 16);   // K = k0+kb+16 .. +23
        fb.u[0] = *(const uint4*)(brow + k0);
        fb.u[1] = *(const uint4*)(brow + k0 + 16);
        acc = __builtin_amdgcn_wmma_f32_16x16x32_bf16(
            /*neg_a=*/false, fa.v, /*neg_b=*/false, fb.v,
            /*c_mod=*/(short)0, acc, /*reuse_a=*/false, /*reuse_b=*/false);
    }

    // C/D layout: VGPR q -> row m0 + q + 8*(lane>=16), col n0 + (lane&15)
    const int crow = m0 + ((lane >> 4) << 3);
    const int ccol = n0 + r;
#pragma unroll
    for (int q = 0; q < 8; ++q)
        D[(long)(crow + q) * ldc + ccol] = acc[q];
}

// ---------------------------------------------------------------------------
// Fused energy + softmax, j-tiled by JT=4. One 256-thread block per (b, j0):
//   e[jj,i] = sum_h V[h] * tanh(We[b,i,h] + Uh[b,j0+jj,h]),  i = 0..511
// JT Uh rows + V staged in LDS; each We element loaded once is reused against
// JT decoder rows (inner mix 1 load : JT tanh -> trans-pipe bound, which is
// the roofline for this op). Then per-row block softmax over 512 energies.
// Writes f32 e to d_out and bf16 e for the context GEMM.
// ---------------------------------------------------------------------------
__global__ __launch_bounds__(256)
void energy_softmax(const float* __restrict__ We, const float* __restrict__ Uh,
                    const float* __restrict__ V,  float* __restrict__ Eout,
                    __bf16* __restrict__ Ebf)
{
    __shared__ float Uh_s[JT][Hd];
    __shared__ float V_s[Hd];
    __shared__ float e_s[JT][TE];
    __shared__ float red_s[8];
    __shared__ float stat_s;

    const int b  = blockIdx.y;
    const int j0 = blockIdx.x * JT;
    const int tid = threadIdx.x, wave = tid >> 5, lane = tid & 31;

#pragma unroll
    for (int jj = 0; jj < JT; ++jj)
        Uh_s[jj][tid] = Uh[((long)b * TD + j0 + jj) * Hd + tid];
    V_s[tid] = V[tid];
    __syncthreads();

    const float* Wb = We + (long)b * TE * Hd;
    for (int i = wave; i < TE; i += 8) {
        const float* wr = Wb + (long)i * Hd;
        float s[JT];
#pragma unroll
        for (int jj = 0; jj < JT; ++jj) s[jj] = 0.f;
#pragma unroll
        for (int t = 0; t < Hd; t += 32) {
            const int h = t + lane;
            const float w = wr[h];           // one coalesced load ...
            const float vh = V_s[h];
#pragma unroll
            for (int jj = 0; jj < JT; ++jj)  // ... feeds JT tanh+fma
                s[jj] = __builtin_fmaf(vh, fast_tanh(w + Uh_s[jj][h]), s[jj]);
        }
#pragma unroll
        for (int jj = 0; jj < JT; ++jj) {
#pragma unroll
            for (int off = 16; off > 0; off >>= 1)
                s[jj] += __shfl_xor(s[jj], off, 32);
            if (lane == 0) e_s[jj][i] = s[jj];
        }
    }
    __syncthreads();

    // per-row softmax over 512 values with 256 threads (2 per thread)
    for (int jj = 0; jj < JT; ++jj) {
        const float v0 = e_s[jj][tid], v1 = e_s[jj][tid + 256];
        float m = fmaxf(v0, v1);
#pragma unroll
        for (int off = 16; off > 0; off >>= 1) m = fmaxf(m, __shfl_xor(m, off, 32));
        if (lane == 0) red_s[wave] = m;
        __syncthreads();
        if (tid == 0) {
            float mm = red_s[0];
            for (int w = 1; w < 8; ++w) mm = fmaxf(mm, red_s[w]);
            stat_s = mm;
        }
        __syncthreads();
        const float gmax = stat_s;

        const float x0 = __expf(v0 - gmax), x1 = __expf(v1 - gmax);
        float ssum = x0 + x1;
#pragma unroll
        for (int off = 16; off > 0; off >>= 1) ssum += __shfl_xor(ssum, off, 32);
        __syncthreads();             // all gmax reads done before smem reuse
        if (lane == 0) red_s[wave] = ssum;
        __syncthreads();
        if (tid == 0) {
            float t = 0.f;
            for (int w = 0; w < 8; ++w) t += red_s[w];
            stat_s = 1.0f / t;
        }
        __syncthreads();
        const float inv = stat_s;

        const long base = ((long)b * TD + j0 + jj) * TE;
        const float o0 = x0 * inv, o1 = x1 * inv;
        Eout[base + tid]       = o0;
        Eout[base + tid + 256] = o1;
        Ebf[base + tid]        = (__bf16)o0;
        Ebf[base + tid + 256]  = (__bf16)o1;
        __syncthreads();             // stat_s/red_s reuse across jj iterations
    }
}

// ---------------------------------------------------------------------------
// Prep kernels: f32 -> bf16 convert, and convert+transpose (per batch slice).
// ---------------------------------------------------------------------------
__global__ void cvt_bf16(const float* __restrict__ X, __bf16* __restrict__ Y, int n)
{
    int i = blockIdx.x * blockDim.x + threadIdx.x;
    if (i < n) Y[i] = (__bf16)X[i];
}

__global__ void transpose_cvt_bf16(const float* __restrict__ X, __bf16* __restrict__ Y,
                                   int rows, int cols)
{
    const long off = (long)blockIdx.z * rows * cols;
    int idx = blockIdx.x * blockDim.x + threadIdx.x;
    if (idx >= rows * cols) return;
    int r = idx / cols, c = idx % cols;
    Y[off + (long)c * rows + r] = (__bf16)X[off + idx];
}

// ---------------------------------------------------------------------------
extern "C" void kernel_launch(void* const* d_in, const int* in_sizes, int n_in,
                              void* d_out, int out_size, void* d_ws, size_t ws_size,
                              hipStream_t stream)
{
    const float* enc = (const float*)d_in[0];  // [B,TE,H]
    const float* dec = (const float*)d_in[1];  // [B,TD,H]
    const float* Wa  = (const float*)d_in[2];  // [H,H]
    const float* Ua  = (const float*)d_in[3];  // [H,H]
    const float* Va  = (const float*)d_in[4];  // [H,1]

    // Workspace carve-out (~14 MB total)
    char* w = (char*)d_ws;
    auto take = [&](size_t bytes) {
        char* p = w;
        w += (bytes + 255) & ~(size_t)255;
        return p;
    };
    __bf16* enc_bf  = (__bf16*)take((size_t)Bsz * TE * Hd * 2); // A for We-GEMM, row-major
    __bf16* encT_bf = (__bf16*)take((size_t)Bsz * TE * Hd * 2); // [b][H][TE] for c-GEMM B
    __bf16* dec_bf  = (__bf16*)take((size_t)Bsz * TD * Hd * 2);
    __bf16* WaT     = (__bf16*)take((size_t)Hd * Hd * 2);       // W_a^T
    __bf16* UaT     = (__bf16*)take((size_t)Hd * Hd * 2);       // U_a^T
    float*  We      = (float*)take((size_t)Bsz * TE * Hd * 4);
    float*  Uh      = (float*)take((size_t)Bsz * TD * Hd * 4);
    __bf16* Ebf     = (__bf16*)take((size_t)Bsz * TD * TE * 2);

    float* c_out = (float*)d_out;                       // [B,TD,H]
    float* e_out = c_out + (size_t)Bsz * TD * Hd;       // [B,TD,TE]

    // 1) precision conversion / operand transposes
    cvt_bf16<<<(Bsz * TE * Hd + 255) / 256, 256, 0, stream>>>(enc, enc_bf, Bsz * TE * Hd);
    cvt_bf16<<<(Bsz * TD * Hd + 255) / 256, 256, 0, stream>>>(dec, dec_bf, Bsz * TD * Hd);
    transpose_cvt_bf16<<<dim3((TE * Hd + 255) / 256, 1, Bsz), 256, 0, stream>>>(enc, encT_bf, TE, Hd);
    transpose_cvt_bf16<<<dim3((Hd * Hd + 255) / 256, 1, 1), 256, 0, stream>>>(Wa, WaT, Hd, Hd);
    transpose_cvt_bf16<<<dim3((Hd * Hd + 255) / 256, 1, 1), 256, 0, stream>>>(Ua, UaT, Hd, Hd);

    // 2) We = enc @ W_a   (M=4096, N=256, K=256)
    gemm_bf16_wmma<<<dim3((Bsz * TE) / 16, Hd / 64, 1), 128, 0, stream>>>(
        enc_bf, WaT, We, Bsz * TE, Hd, Hd, Hd, Hd, Hd, 0, 0, 0);

    // 3) Uh = dec @ U_a   (M=2048, N=256, K=256)
    gemm_bf16_wmma<<<dim3((Bsz * TD) / 16, Hd / 64, 1), 128, 0, stream>>>(
        dec_bf, UaT, Uh, Bsz * TD, Hd, Hd, Hd, Hd, Hd, 0, 0, 0);

    // 4) fused energies + softmax -> e (f32 output) and e (bf16 for context GEMM)
    energy_softmax<<<dim3(TD / JT, Bsz), 256, 0, stream>>>(We, Uh, Va, e_out, Ebf);

    // 5) c[b] = e[b] @ enc[b]  (batched, M=256, N=256, K=512)
    gemm_bf16_wmma<<<dim3(TD / 16, Hd / 64, Bsz), 128, 0, stream>>>(
        Ebf, encT_bf, c_out, TD, Hd, TE, TE, TE, Hd,
        (long)TD * TE, (long)Hd * TE, (long)TD * Hd);
}
